// GroupedQueryAttention_51427938402923
// MI455X (gfx1250) — matrix operand order repaired and checked by reading, not compiled, not run on hardware
//
#include <hip/hip_runtime.h>

typedef _Float16 half_t;
typedef __attribute__((ext_vector_type(2)))  _Float16 h2;
typedef __attribute__((ext_vector_type(16))) _Float16 v16h;
typedef __attribute__((ext_vector_type(8)))  float    v8f;
typedef __attribute__((ext_vector_type(4)))  unsigned int u32x4;
typedef __attribute__((ext_vector_type(8)))  int      i32x8;
typedef __attribute__((ext_vector_type(4)))  int      i32x4;

#define DIMC     512
#define HEADS    8
#define HEAD_DIM 64
#define BATCH    2
#define SEQ      3136
#define OC3      1536
#define ATT_SCALE 0.125f   // 64^-0.5

// ---------------------------------------------------------------------------
// WMMA fragment helpers (layouts per CDNA5 ISA 7.12.2, wave32)
// A 16x32 f16: lane (g=l/16, m=l%16) holds row m; VGPR i pairs
//   K = (i<4?0:16) + 8*g + 2*(i&3) .. +1
// B 32x16 f16: lane (g, n) holds column n; VGPR i pairs
//   K = (i<4?0:16) + 8*g + 2*(i&3) .. +1
// C/D 16x16 f32: VGPR i -> row (i + 8*g), col n
// ---------------------------------------------------------------------------

static __device__ __forceinline__ v16h load_frag_a(const half_t* __restrict__ A,
                                                   int lda, int row0, int k0, int lane) {
  const int g = lane >> 4, m = lane & 15;
  const half_t* p = A + (size_t)(row0 + m) * lda + k0;
  v16h f;
#pragma unroll
  for (int i = 0; i < 8; ++i) {
    const int kk = ((i < 4) ? 0 : 16) + g * 8 + (i & 3) * 2;
    h2 pr = *(const h2*)(p + kk);
    f[2 * i] = pr[0];
    f[2 * i + 1] = pr[1];
  }
  return f;
}

// Bt is "column-major B": element B[k][n] stored at Bt[n*ldb + k]
static __device__ __forceinline__ v16h load_frag_b(const half_t* __restrict__ Bt,
                                                   int ldb, int k0, int col0, int lane) {
  const int g = lane >> 4, n = lane & 15;
  const half_t* p = Bt + (size_t)(col0 + n) * ldb + k0;
  v16h f;
#pragma unroll
  for (int i = 0; i < 8; ++i) {
    const int kk = ((i < 4) ? 0 : 16) + g * 8 + (i & 3) * 2;
    h2 pr = *(const h2*)(p + kk);
    f[2 * i] = pr[0];
    f[2 * i + 1] = pr[1];
  }
  return f;
}

static __device__ __forceinline__ v8f wmma16(v16h a, v16h b, v8f c) {
  return __builtin_amdgcn_wmma_f32_16x16x32_f16(false, a, false, b, (short)0, c,
                                                false, false);
}

// ---------------------------------------------------------------------------
// Tensor Data Mover: 2D tile load global -> LDS (CDNA5 ISA ch.8 descriptor).
// 2-byte elements. group0: count=1, lds byte addr, 57b global addr, type=2.
// group1: data_size=1(2B); tensor_dim0/1, tile_dim0/1, tensor_dim0_stride.
// Remaining groups zero (2D tensor). Issued by one wave; tracked by TENSORcnt.
// ---------------------------------------------------------------------------
static __device__ __forceinline__ void tdm_load_2d_f16(unsigned lds_byte_off,
                                                       const void* gaddr,
                                                       unsigned tensor_dim0,
                                                       unsigned tensor_dim1,
                                                       unsigned tile_dim0,
                                                       unsigned tile_dim1,
                                                       unsigned stride0) {
  unsigned long long ga = (unsigned long long)gaddr;
  u32x4 g0;
  g0[0] = 1u;                                   // count=1, user mode
  g0[1] = lds_byte_off;                         // lds_addr
  g0[2] = (unsigned)ga;                         // global_addr[31:0]
  g0[3] = (unsigned)((ga >> 32) & 0x01FFFFFFu) | 0x80000000u;  // addr[56:32] | type=2
  i32x8 g1;
  g1[0] = (int)(1u << 16);                      // data_size = 2 bytes
  g1[1] = (int)(tensor_dim0 << 16);             // tensor_dim0[15:0]
  g1[2] = (int)((tensor_dim0 >> 16) | (tensor_dim1 << 16));
  g1[3] = (int)((tensor_dim1 >> 16) | (tile_dim0 << 16));
  g1[4] = (int)(tile_dim1 & 0xffffu);           // tile_dim1, tile_dim2=0
  g1[5] = (int)stride0;                         // tensor_dim0_stride[31:0]
  g1[6] = 0;
  g1[7] = 0;
  i32x4 gz4;
  gz4[0] = 0; gz4[1] = 0; gz4[2] = 0; gz4[3] = 0;
  i32x8 gz8;
  gz8[0] = 0; gz8[1] = 0; gz8[2] = 0; gz8[3] = 0;
  gz8[4] = 0; gz8[5] = 0; gz8[6] = 0; gz8[7] = 0;
  __builtin_amdgcn_tensor_load_to_lds(g0, g1, gz4, gz4, gz8, 0);
}

// ---------------------------------------------------------------------------
// Pre-pass conversions
// ---------------------------------------------------------------------------

__global__ __launch_bounds__(256) void cvt_f32_to_f16(const float* __restrict__ src,
                                                      half_t* __restrict__ dst, int n) {
  int i = blockIdx.x * blockDim.x + threadIdx.x;
  if (i < n) dst[i] = (half_t)src[i];
}

// x: [B][C][S] f32  ->  xt: [B][S][C] f16
__global__ __launch_bounds__(256) void transpose_x_f16(const float* __restrict__ x,
                                                       half_t* __restrict__ xt) {
  int i = blockIdx.x * blockDim.x + threadIdx.x;
  const int CS = DIMC * SEQ;
  if (i >= BATCH * CS) return;
  int b = i / CS, r = i - b * CS;
  int c = r / SEQ, s = r - c * SEQ;
  xt[((size_t)b * SEQ + s) * DIMC + c] = (half_t)x[i];
}

// ---------------------------------------------------------------------------
// Software-pipelined 32x64-per-wave WMMA GEMM core: fragments for step k+1
// are issued before the WMMAs of step k so loads stay in flight under compute.
// ---------------------------------------------------------------------------
static __device__ __forceinline__ void gemm_core_32x64(const half_t* __restrict__ A,
                                                       const half_t* __restrict__ Bt,
                                                       int mBase, int nBase, int lane,
                                                       v8f acc[2][4]) {
  v16h a0 = load_frag_a(A, DIMC, mBase, 0, lane);
  v16h a1 = load_frag_a(A, DIMC, mBase + 16, 0, lane);
  v16h bf[4];
#pragma unroll
  for (int nt = 0; nt < 4; ++nt) bf[nt] = load_frag_b(Bt, DIMC, 0, nBase + nt * 16, lane);

  for (int k0 = 0; k0 < DIMC; k0 += 32) {
    const int kn = k0 + 32;
    v16h na0, na1, nbf[4];
    if (kn < DIMC) {
      __builtin_prefetch(A + (size_t)(mBase + (lane & 15)) * DIMC + kn + 32);
      __builtin_prefetch(Bt + (size_t)(nBase + (lane & 15)) * DIMC + kn + 32);
      na0 = load_frag_a(A, DIMC, mBase, kn, lane);
      na1 = load_frag_a(A, DIMC, mBase + 16, kn, lane);
#pragma unroll
      for (int nt = 0; nt < 4; ++nt)
        nbf[nt] = load_frag_b(Bt, DIMC, kn, nBase + nt * 16, lane);
    }
#pragma unroll
    for (int nt = 0; nt < 4; ++nt) {
      acc[0][nt] = wmma16(a0, bf[nt], acc[0][nt]);
      acc[1][nt] = wmma16(a1, bf[nt], acc[1][nt]);
    }
    if (kn < DIMC) {
      a0 = na0;
      a1 = na1;
#pragma unroll
      for (int nt = 0; nt < 4; ++nt) bf[nt] = nbf[nt];
    }
  }
}

// ---------------------------------------------------------------------------
// QKV GEMM: [1536 x 512] @ [512 x 3136] per batch.
// Epilogue scatters into q[b,h,s,d], k[b,h,s,d], v^T[b,h,d,s] (all f16).
// ---------------------------------------------------------------------------

__global__ __launch_bounds__(32) void qkv_gemm(const half_t* __restrict__ W,
                                               const half_t* __restrict__ Xt,
                                               half_t* __restrict__ Q,
                                               half_t* __restrict__ Kh,
                                               half_t* __restrict__ Vt) {
  const int lane = threadIdx.x;
  const int nBase = blockIdx.x * 64;
  const int mBase = blockIdx.y * 32;
  const int b = blockIdx.z;
  const half_t* Bt = Xt + (size_t)b * SEQ * DIMC;

  v8f acc[2][4] = {};
  gemm_core_32x64(W, Bt, mBase, nBase, lane, acc);

  const int g = lane >> 4, n = lane & 15;
#pragma unroll
  for (int mt = 0; mt < 2; ++mt)
#pragma unroll
    for (int nt = 0; nt < 4; ++nt)
#pragma unroll
      for (int i = 0; i < 8; ++i) {
        int o = mBase + mt * 16 + i + 8 * g;
        int s = nBase + nt * 16 + n;
        float v = acc[mt][nt][i];
        int which = o >> 9;            // 0=q 1=k 2=v
        int rem = o & 511;
        int hh = rem >> 6;
        int d = rem & 63;
        size_t bh = (size_t)b * HEADS + hh;
        if (which == 0)
          Q[(bh * SEQ + s) * HEAD_DIM + d] = (half_t)v;
        else if (which == 1)
          Kh[(bh * SEQ + s) * HEAD_DIM + d] = (half_t)v;
        else
          Vt[(bh * HEAD_DIM + d) * SEQ + s] = (half_t)v;
      }
}

// ---------------------------------------------------------------------------
// L2 normalize rows of 64 f16 values (one wave per row, 2 elems/lane).
// ---------------------------------------------------------------------------

__global__ __launch_bounds__(256) void l2norm_rows(half_t* __restrict__ T, int rows) {
  int lane = threadIdx.x & 31;
  int wave = blockIdx.x * (blockDim.x >> 5) + (threadIdx.x >> 5);
  int nw = gridDim.x * (blockDim.x >> 5);
  for (int r = wave; r < rows; r += nw) {
    half_t* p = T + (size_t)r * HEAD_DIM + lane * 2;
    h2 v = *(h2*)p;
    float a = (float)v[0], c = (float)v[1];
    float ss = a * a + c * c;
#pragma unroll
    for (int m = 16; m >= 1; m >>= 1) ss += __shfl_xor(ss, m, 32);
    float sc = 1.0f / fmaxf(sqrtf(ss), 1e-12f);
    v[0] = (half_t)(a * sc);
    v[1] = (half_t)(c * sc);
    *(h2*)p = v;
  }
}

// ---------------------------------------------------------------------------
// Flash attention v3: 4 waves / workgroup, 64 q rows per block (16 per wave).
// Double-buffered TDM: wave 0 issues the next chunk's two tensor loads into
// the alternate LDS buffer, then s_wait_tensorcnt(2) — TDM ops from one wave
// complete in order, so <=2 outstanding guarantees the current chunk landed
// while the next streams in under QK^T (WMMA) + softmax + PV (WMMA).
// Output written transposed: AOt[b][s][c] f16 (proj GEMM B-matrix layout).
// ---------------------------------------------------------------------------

#define SK_ELEMS  (32 * HEAD_DIM)        // 2048 f16 = 4KB
#define SV_ELEMS  (HEAD_DIM * 32)        // 2048 f16 = 4KB
#define BUF_ELEMS (SK_ELEMS + SV_ELEMS)  // 8KB per buffer
#define SP_ELEMS  (16 * 32)              // per-wave P tile

__global__ __launch_bounds__(128) void flash_attn(const half_t* __restrict__ Q,
                                                  const half_t* __restrict__ Kk,
                                                  const half_t* __restrict__ Vt,
                                                  half_t* __restrict__ AOt) {
  __shared__ __align__(16) half_t smem[2 * BUF_ELEMS + 4 * SP_ELEMS];
  const int w = threadIdx.x >> 5;
  const int lane = threadIdx.x & 31;
  half_t* sP = smem + 2 * BUF_ELEMS + w * SP_ELEMS;

  const int g = lane >> 4, n = lane & 15;
  const int s0 = blockIdx.x * 64 + w * 16;
  const int h = blockIdx.y;
  const int b = blockIdx.z;
  const size_t bh = (size_t)b * HEADS + h;
  const half_t* Qh = Q + bh * SEQ * HEAD_DIM;   // [s][d]
  const half_t* Kh = Kk + bh * SEQ * HEAD_DIM;  // [t][d]
  const half_t* Vh = Vt + bh * HEAD_DIM * SEQ;  // [d][t]

  v16h qa[2];
#pragma unroll
  for (int kc = 0; kc < 2; ++kc) qa[kc] = load_frag_a(Qh, HEAD_DIM, s0, kc * 32, lane);

  v8f accO[4] = {};
  float mrow[8], lrow[8];
#pragma unroll
  for (int i = 0; i < 8; ++i) { mrow[i] = -1e30f; lrow[i] = 0.f; }

  // prologue: DMA chunk 0 into buffer 0
  if (w == 0) {
    tdm_load_2d_f16(0u, Kh, HEAD_DIM, SEQ, HEAD_DIM, 32, HEAD_DIM);
    tdm_load_2d_f16((unsigned)(SK_ELEMS * sizeof(half_t)), Vh,
                    SEQ, HEAD_DIM, 32, HEAD_DIM, SEQ);
  }

  for (int tt = 0; tt < SEQ; tt += 32) {
    const int cur = (tt >> 5) & 1;
    half_t* sK = smem + cur * BUF_ELEMS;      // [32 t][64 d]
    half_t* sV = sK + SK_ELEMS;               // [64 d][32 t]

    if (w == 0) {
      if (tt + 32 < SEQ) {
        const unsigned noff = (unsigned)((cur ^ 1) * BUF_ELEMS * sizeof(half_t));
        tdm_load_2d_f16(noff, Kh + (size_t)(tt + 32) * HEAD_DIM,
                        HEAD_DIM, SEQ, HEAD_DIM, 32, HEAD_DIM);
        tdm_load_2d_f16(noff + (unsigned)(SK_ELEMS * sizeof(half_t)), Vh + tt + 32,
                        SEQ, HEAD_DIM, 32, HEAD_DIM, SEQ);
        __builtin_amdgcn_s_wait_tensorcnt(2);  // current chunk landed
      } else {
        __builtin_amdgcn_s_wait_tensorcnt(0);
      }
    }
    __syncthreads();   // current-chunk LDS tiles visible to all waves

    // scores: two 16-col tiles, K-reduction over head_dim = 64
    v8f sc[2];
#pragma unroll
    for (int ct = 0; ct < 2; ++ct) {
      v8f z = {};
#pragma unroll
      for (int kc = 0; kc < 2; ++kc) {
        v16h kb = load_frag_b(sK, HEAD_DIM, kc * 32, ct * 16, lane);
        z = wmma16(qa[kc], kb, z);
      }
      sc[ct] = z;
    }

    // online softmax over the 32 new columns
    float alpha[8];
#pragma unroll
    for (int i = 0; i < 8; ++i) {
      float x0 = sc[0][i] * ATT_SCALE;
      float x1 = sc[1][i] * ATT_SCALE;
      float mx = fmaxf(x0, x1);
#pragma unroll
      for (int msk = 8; msk >= 1; msk >>= 1) mx = fmaxf(mx, __shfl_xor(mx, msk, 32));
      float mn = fmaxf(mrow[i], mx);
      alpha[i] = __expf(mrow[i] - mn);
      mrow[i] = mn;
      float p0 = __expf(x0 - mn);
      float p1 = __expf(x1 - mn);
      sc[0][i] = p0;
      sc[1][i] = p1;
      float rs = p0 + p1;
#pragma unroll
      for (int msk = 8; msk >= 1; msk >>= 1) rs += __shfl_xor(rs, msk, 32);
      lrow[i] = lrow[i] * alpha[i] + rs;
    }

    // rescale running output accumulators (same row<->VGPR mapping)
#pragma unroll
    for (int dt = 0; dt < 4; ++dt)
#pragma unroll
      for (int i = 0; i < 8; ++i) accO[dt][i] *= alpha[i];

    // stage P (16x32, f16) through per-wave LDS: D-layout -> A-layout
#pragma unroll
    for (int ct = 0; ct < 2; ++ct)
#pragma unroll
      for (int i = 0; i < 8; ++i)
        sP[(i + 8 * g) * 32 + ct * 16 + n] = (half_t)sc[ct][i];

    v16h pa = load_frag_a(sP, 32, 0, 0, lane);

    // PV: accumulate over this 32-t chunk into 4 d-tiles
#pragma unroll
    for (int dt = 0; dt < 4; ++dt) {
      v16h vb = load_frag_b(sV, 32, 0, dt * 16, lane);
      accO[dt] = wmma16(pa, vb, accO[dt]);
    }

    __syncthreads();   // buffer consumed; safe target for next iteration's DMA
  }

  // epilogue: out[b][s][h*64 + d] f16
#pragma unroll
  for (int dt = 0; dt < 4; ++dt)
#pragma unroll
    for (int i = 0; i < 8; ++i) {
      float v = accO[dt][i] / lrow[i];
      int s = s0 + i + 8 * g;
      AOt[((size_t)b * SEQ + s) * DIMC + h * HEAD_DIM + dt * 16 + n] = (half_t)v;
    }
}

// ---------------------------------------------------------------------------
// Output projection: [512 x 512] @ [512 x 3136] + bias, f32 out [b][o][s].
// ---------------------------------------------------------------------------

__global__ __launch_bounds__(32) void proj_gemm(const half_t* __restrict__ W,
                                                const half_t* __restrict__ At,
                                                const float* __restrict__ bias,
                                                float* __restrict__ Out) {
  const int lane = threadIdx.x;
  const int nBase = blockIdx.x * 64;
  const int mBase = blockIdx.y * 32;
  const int b = blockIdx.z;
  const half_t* Bt = At + (size_t)b * SEQ * DIMC;

  v8f acc[2][4] = {};
  gemm_core_32x64(W, Bt, mBase, nBase, lane, acc);

  const int g = lane >> 4, n = lane & 15;
#pragma unroll
  for (int mt = 0; mt < 2; ++mt)
#pragma unroll
    for (int nt = 0; nt < 4; ++nt)
#pragma unroll
      for (int i = 0; i < 8; ++i) {
        int o = mBase + mt * 16 + i + 8 * g;
        int s = nBase + nt * 16 + n;
        Out[((size_t)b * DIMC + o) * SEQ + s] = acc[mt][nt][i] + bias[o];
      }
}

// ---------------------------------------------------------------------------
// Launch
// ---------------------------------------------------------------------------

extern "C" void kernel_launch(void* const* d_in, const int* in_sizes, int n_in,
                              void* d_out, int out_size, void* d_ws, size_t ws_size,
                              hipStream_t stream) {
  (void)in_sizes; (void)n_in; (void)out_size; (void)ws_size;
  const float* x      = (const float*)d_in[0];
  const float* w_qkv  = (const float*)d_in[1];
  const float* w_proj = (const float*)d_in[2];
  const float* b_proj = (const float*)d_in[3];
  float* out = (float*)d_out;

  half_t* ws = (half_t*)d_ws;
  const size_t N_XT = (size_t)BATCH * SEQ * DIMC;              // 3,211,264
  const size_t N_WQ = (size_t)OC3 * DIMC;                      //   786,432
  const size_t N_WP = (size_t)DIMC * DIMC;                     //   262,144
  const size_t N_T  = (size_t)BATCH * HEADS * SEQ * HEAD_DIM;  // 3,211,264

  half_t* x16t    = ws;
  half_t* wqkv16  = x16t + N_XT;
  half_t* wproj16 = wqkv16 + N_WQ;
  half_t* q16     = wproj16 + N_WP;
  half_t* k16     = q16 + N_T;
  half_t* v16t    = k16 + N_T;
  half_t* ao16t   = v16t + N_T;

  cvt_f32_to_f16<<<(int)((N_WQ + 255) / 256), 256, 0, stream>>>(w_qkv, wqkv16, (int)N_WQ);
  cvt_f32_to_f16<<<(int)((N_WP + 255) / 256), 256, 0, stream>>>(w_proj, wproj16, (int)N_WP);
  transpose_x_f16<<<(int)((N_XT + 255) / 256), 256, 0, stream>>>(x, x16t);

  qkv_gemm<<<dim3(SEQ / 64, OC3 / 32, BATCH), 32, 0, stream>>>(wqkv16, x16t, q16, k16, v16t);

  const int rows = BATCH * HEADS * SEQ;
  l2norm_rows<<<256, 256, 0, stream>>>(q16, rows);
  l2norm_rows<<<256, 256, 0, stream>>>(k16, rows);

  flash_attn<<<dim3(SEQ / 64, HEADS, BATCH), 128, 0, stream>>>(q16, k16, v16t, ao16t);

  proj_gemm<<<dim3(SEQ / 64, DIMC / 32, BATCH), 32, 0, stream>>>(wproj16, ao16t, b_proj, out);
}
